// DGLEnGraphConv_10196252360942
// MI455X (gfx1250) — matrix operand order repaired and checked by reading, not compiled, not run on hardware
//
#include <hip/hip_runtime.h>
#include <hip/hip_bf16.h>

typedef __attribute__((ext_vector_type(16))) _Float16 v16h;
typedef __attribute__((ext_vector_type(8)))  float    v8f;

#define N_NODES 50000
#define N_EDGES 800000

// padded LDS row strides (halves): word-stride mod 64 banks = 36 / 4 / 4 -> conflict-free row lanes
#define LDA_E 328   // edge m_in rows (K logical 320)
#define LDA_N 264   // node concat rows (K logical 256)
#define LDM   136   // 128-wide intermediate rows

union FragU { v16h v; uint4 q[2]; };
union Pack4 { _Float16 h[4]; uint2 u; };

__device__ __forceinline__ float silu_f(float x) { return x / (1.0f + __expf(-x)); }
__device__ __forceinline__ float sigm_f(float x) { return 1.0f / (1.0f + __expf(-x)); }

// A fragment (16xK tile in LDS, row-major, ldk halves/row).
// Lane L: M = L%16; halves 0..7 -> K = kt*32 + (L>=16?8:0) + h ; halves 8..15 -> +16.
__device__ __forceinline__ v16h load_a_frag(const _Float16* Abase, int ldk, int kt, int lane) {
  int row = lane & 15;
  int ks  = kt * 32 + ((lane >> 4) << 3);
  const _Float16* p = Abase + row * ldk + ks;
  FragU f;
  f.q[0] = *(const uint4*)(p);
  f.q[1] = *(const uint4*)(p + 16);
  return f.v;
}

// B fragment from pre-permuted weights: fragment (kt,nt) = 32 lanes x 16 halves contiguous.
__device__ __forceinline__ v16h load_b_frag(const _Float16* Wp, int kt, int nt, int lane) {
  const _Float16* p = Wp + (((size_t)(kt * 8 + nt)) * 512 + lane * 16);
  FragU f;
  f.q[0] = *(const uint4*)(p);
  f.q[1] = *(const uint4*)(p + 8);
  return f.v;
}

__global__ void zero_kernel(float* __restrict__ p, int n) {
  for (int i = blockIdx.x * blockDim.x + threadIdx.x; i < n; i += gridDim.x * blockDim.x)
    p[i] = 0.0f;
}

// Permute W[K x 128] (row-major f32) into f16 B-fragments.
// grid.x = KT*8, block = 512. (lane,h) of frag (kt,nt) = W[kt*32 + (h%8) + 16*(h>=8) + 8*(lane>=16)][nt*16 + lane%16]
__global__ void permute_w(const float* __restrict__ W, int Kact, _Float16* __restrict__ out) {
  int b = blockIdx.x;
  int kt = b >> 3, nt = b & 7;
  int t = threadIdx.x;
  int lane = t >> 4, h = t & 15;
  int k = kt * 32 + (h & 7) + ((h >> 3) << 4) + ((lane >> 4) << 3);
  int n = nt * 16 + (lane & 15);
  float v = (k < Kact) ? W[(size_t)k * 128 + n] : 0.0f;
  out[(size_t)b * 512 + lane * 16 + h] = (_Float16)v;
}

// ---------------- Edge kernel: 2 waves/block, 32 edges/wave (2 M-tiles), 64 edges/block ----------------
__launch_bounds__(64)
__global__ void egnn_edge_kernel(
    const float* __restrict__ nf, const float* __restrict__ coords,
    const float* __restrict__ ef,
    const float* __restrict__ b_e1, const float* __restrict__ b_e2,
    const float* __restrict__ W_a, const float* __restrict__ b_a,
    const float* __restrict__ b_c1, const float* __restrict__ W_c,
    const _Float16* __restrict__ We1p, const _Float16* __restrict__ We2p,
    const _Float16* __restrict__ Wc1p,
    const int* __restrict__ src, const int* __restrict__ dst,
    float* __restrict__ m_agg, float* __restrict__ coord_agg, float* __restrict__ cnt)
{
  __shared__ _Float16 sA[2][32 * LDA_E];   // m_in staging; later reused for gated m (ld=LDM)
  __shared__ _Float16 sM1[2][32 * LDM];    // silu(GEMM1); reused as f32 gate scratch after GEMM2
  __shared__ float sCD[64][3];
  __shared__ float sG[64];
  __shared__ float sC[64];
  __shared__ int   sDst[64];

  const int tid  = threadIdx.x;
  const int wave = tid >> 5;
  const int lane = tid & 31;
  const int ewave = blockIdx.x * 64 + wave * 32;

  _Float16* A  = sA[wave];
  _Float16* M1 = sM1[wave];

  // ---- Stage A = [nf[src] | nf[dst] | radial | edge_feats | 0-pad]; 1 lane per edge ----
  {
    const int e = ewave + lane;
    _Float16* Arow = A + lane * LDA_E;
    if (e < N_EDGES) {
      int s = src[e], d = dst[e];
      const float4* rs = (const float4*)(nf + (size_t)s * 128);
      const float4* rd = (const float4*)(nf + (size_t)d * 128);
      #pragma unroll 8
      for (int j = 0; j < 32; ++j) {
        float4 f = rs[j];
        Pack4 p; p.h[0] = (_Float16)f.x; p.h[1] = (_Float16)f.y;
                 p.h[2] = (_Float16)f.z; p.h[3] = (_Float16)f.w;
        *(uint2*)(Arow + 4 * j) = p.u;
      }
      #pragma unroll 8
      for (int j = 0; j < 32; ++j) {
        float4 f = rd[j];
        Pack4 p; p.h[0] = (_Float16)f.x; p.h[1] = (_Float16)f.y;
                 p.h[2] = (_Float16)f.z; p.h[3] = (_Float16)f.w;
        *(uint2*)(Arow + 128 + 4 * j) = p.u;
      }
      float cx = coords[3*s+0] - coords[3*d+0];
      float cy = coords[3*s+1] - coords[3*d+1];
      float cz = coords[3*s+2] - coords[3*d+2];
      sCD[wave*32+lane][0] = cx; sCD[wave*32+lane][1] = cy; sCD[wave*32+lane][2] = cz;
      sDst[wave*32+lane] = d;
      Arow[256] = (_Float16)(cx*cx + cy*cy + cz*cz);
      const float* er = ef + (size_t)e * 32;
      for (int j = 0; j < 32; ++j) Arow[257 + j] = (_Float16)er[j];
      for (int j = 289; j < 320; ++j) Arow[j] = (_Float16)0.0f;
    } else {
      for (int j = 0; j < 320; ++j) Arow[j] = (_Float16)0.0f;
      sDst[wave*32+lane] = 0;
    }
  }
  __syncthreads();

  const int row0 = (lane >> 4) << 3;   // D-frag: M = v + 8*(lane>=16)
  const int ncol = lane & 15;

  // ---- GEMM1: m1 = silu(m_in @ W_e1 + b_e1)   (K = 320, 2 M-tiles share each B frag) ----
  for (int nt = 0; nt < 8; ++nt) {
    float bv = b_e1[nt * 16 + ncol];
    v8f acc0, acc1;
    for (int v = 0; v < 8; ++v) { acc0[v] = bv; acc1[v] = bv; }
    for (int kt = 0; kt < 10; ++kt) {
      v16h b  = load_b_frag(We1p, kt, nt, lane);
      v16h a0 = load_a_frag(A, LDA_E, kt, lane);
      v16h a1 = load_a_frag(A + 16 * LDA_E, LDA_E, kt, lane);
      acc0 = __builtin_amdgcn_wmma_f32_16x16x32_f16(false, a0, false, b, (short)0, acc0, false, false);
      acc1 = __builtin_amdgcn_wmma_f32_16x16x32_f16(false, a1, false, b, (short)0, acc1, false, false);
    }
    int col = nt * 16 + ncol;
    for (int v = 0; v < 8; ++v) {
      M1[(row0 + v) * LDM + col]        = (_Float16)silu_f(acc0[v]);
      M1[(16 + row0 + v) * LDM + col]   = (_Float16)silu_f(acc1[v]);
    }
  }
  __syncthreads();

  // ---- GEMM2: m2 = silu(m1 @ W_e2 + b_e2)  (K = 128); store into A region (dead), ld=LDM ----
  _Float16* M2 = A;
  for (int nt = 0; nt < 8; ++nt) {
    float bv = b_e2[nt * 16 + ncol];
    v8f acc0, acc1;
    for (int v = 0; v < 8; ++v) { acc0[v] = bv; acc1[v] = bv; }
    for (int kt = 0; kt < 4; ++kt) {
      v16h b  = load_b_frag(We2p, kt, nt, lane);
      v16h a0 = load_a_frag(M1, LDM, kt, lane);
      v16h a1 = load_a_frag(M1 + 16 * LDM, LDM, kt, lane);
      acc0 = __builtin_amdgcn_wmma_f32_16x16x32_f16(false, a0, false, b, (short)0, acc0, false, false);
      acc1 = __builtin_amdgcn_wmma_f32_16x16x32_f16(false, a1, false, b, (short)0, acc1, false, false);
    }
    int col = nt * 16 + ncol;
    for (int v = 0; v < 8; ++v) {
      M2[(row0 + v) * LDM + col]        = (_Float16)silu_f(acc0[v]);
      M2[(16 + row0 + v) * LDM + col]   = (_Float16)silu_f(acc1[v]);
    }
  }
  __syncthreads();

  // ---- Attention gate: m2 *= sigmoid(m2 @ W_a + b_a) ----
  // Phase 1: lane owns a 4-col strip across all 32 edges (conflict-free column reads)
  {
    float* sT = (float*)M1;          // M1 dead; 32x33 f32 scratch
    float wa0 = W_a[lane*4+0], wa1 = W_a[lane*4+1], wa2 = W_a[lane*4+2], wa3 = W_a[lane*4+3];
    for (int e = 0; e < 32; ++e) {
      const _Float16* p = M2 + e * LDM + lane * 4;
      Pack4 q; q.u = *(const uint2*)p;
      sT[e * 33 + lane] = (float)q.h[0]*wa0 + (float)q.h[1]*wa1 + (float)q.h[2]*wa2 + (float)q.h[3]*wa3;
    }
  }
  __syncthreads();
  {
    float* sT = (float*)M1;
    float s = 0.0f;
    for (int l = 0; l < 32; ++l) s += sT[lane * 33 + l];
    sG[wave * 32 + lane] = sigm_f(s + b_a[0]);
  }
  __syncthreads();
  for (int idx = lane; idx < 32 * 128; idx += 32) {
    int el = idx >> 7, col = idx & 127;
    M2[el * LDM + col] = (_Float16)((float)M2[el * LDM + col] * sG[wave * 32 + el]);
  }
  __syncthreads();

  // ---- GEMM3: c = silu(m2 @ W_c1 + b_c1) @ W_c   (fused reduction over N) ----
  {
    float part0[8], part1[8];
    for (int v = 0; v < 8; ++v) { part0[v] = 0.0f; part1[v] = 0.0f; }
    for (int nt = 0; nt < 8; ++nt) {
      float bv = b_c1[nt * 16 + ncol];
      v8f acc0, acc1;
      for (int v = 0; v < 8; ++v) { acc0[v] = bv; acc1[v] = bv; }
      for (int kt = 0; kt < 4; ++kt) {
        v16h b  = load_b_frag(Wc1p, kt, nt, lane);
        v16h a0 = load_a_frag(M2, LDM, kt, lane);
        v16h a1 = load_a_frag(M2 + 16 * LDM, LDM, kt, lane);
        acc0 = __builtin_amdgcn_wmma_f32_16x16x32_f16(false, a0, false, b, (short)0, acc0, false, false);
        acc1 = __builtin_amdgcn_wmma_f32_16x16x32_f16(false, a1, false, b, (short)0, acc1, false, false);
      }
      float wc = W_c[nt * 16 + ncol];
      for (int v = 0; v < 8; ++v) {
        part0[v] += silu_f(acc0[v]) * wc;
        part1[v] += silu_f(acc1[v]) * wc;
      }
    }
    for (int off = 1; off < 16; off <<= 1)
      for (int v = 0; v < 8; ++v) {
        part0[v] += __shfl_xor(part0[v], off, 32);
        part1[v] += __shfl_xor(part1[v], off, 32);
      }
    if ((lane & 15) == 0)
      for (int v = 0; v < 8; ++v) {
        sC[wave * 32 + row0 + v]      = part0[v];
        sC[wave * 32 + 16 + row0 + v] = part1[v];
      }
  }
  __syncthreads();

  // ---- Aggregation: segment sums via global f32 atomics ----
  for (int idx = lane; idx < 32 * 128; idx += 32) {
    int el = idx >> 7, col = idx & 127;
    int e = ewave + el;
    if (e < N_EDGES) {
      int d = sDst[wave * 32 + el];
      atomicAdd(&m_agg[(size_t)d * 128 + col], (float)M2[el * LDM + col]);
    }
  }
  {
    int e = ewave + lane;
    if (e < N_EDGES) {
      int d = sDst[wave * 32 + lane];
      float c = sC[wave * 32 + lane];
      atomicAdd(&coord_agg[3 * d + 0], sCD[wave * 32 + lane][0] * c);
      atomicAdd(&coord_agg[3 * d + 1], sCD[wave * 32 + lane][1] * c);
      atomicAdd(&coord_agg[3 * d + 2], sCD[wave * 32 + lane][2] * c);
      atomicAdd(&cnt[d], 1.0f);
    }
  }
}

// ---------------- Node kernel: 2 waves/block, 32 nodes/wave ----------------
__launch_bounds__(64)
__global__ void egnn_node_kernel(
    const float* __restrict__ nf, const float* __restrict__ coords,
    const float* __restrict__ b_n1, const float* __restrict__ b_n2,
    const _Float16* __restrict__ Wn1p, const _Float16* __restrict__ Wn2p,
    const float* __restrict__ m_agg, const float* __restrict__ coord_agg,
    const float* __restrict__ cnt,
    float* __restrict__ h_out, float* __restrict__ c_out)
{
  __shared__ _Float16 sA[2][32 * LDA_N];
  __shared__ _Float16 sM1[2][32 * LDM];
  const int tid = threadIdx.x, wave = tid >> 5, lane = tid & 31;
  const int nwave = blockIdx.x * 64 + wave * 32;
  _Float16* A  = sA[wave];
  _Float16* M1 = sM1[wave];

  // stage [nf | m_agg]; 1 lane per node
  {
    const int n = nwave + lane;
    _Float16* Arow = A + lane * LDA_N;
    if (n < N_NODES) {
      const float4* r0 = (const float4*)(nf + (size_t)n * 128);
      const float4* r1 = (const float4*)(m_agg + (size_t)n * 128);
      #pragma unroll 8
      for (int j = 0; j < 32; ++j) {
        float4 f = r0[j];
        Pack4 p; p.h[0] = (_Float16)f.x; p.h[1] = (_Float16)f.y;
                 p.h[2] = (_Float16)f.z; p.h[3] = (_Float16)f.w;
        *(uint2*)(Arow + 4 * j) = p.u;
      }
      #pragma unroll 8
      for (int j = 0; j < 32; ++j) {
        float4 f = r1[j];
        Pack4 p; p.h[0] = (_Float16)f.x; p.h[1] = (_Float16)f.y;
                 p.h[2] = (_Float16)f.z; p.h[3] = (_Float16)f.w;
        *(uint2*)(Arow + 128 + 4 * j) = p.u;
      }
    } else {
      for (int j = 0; j < 256; ++j) Arow[j] = (_Float16)0.0f;
    }
  }
  __syncthreads();

  const int row0 = (lane >> 4) << 3;
  const int ncol = lane & 15;

  // GEMM1: K = 256
  for (int nt = 0; nt < 8; ++nt) {
    float bv = b_n1[nt * 16 + ncol];
    v8f acc0, acc1;
    for (int v = 0; v < 8; ++v) { acc0[v] = bv; acc1[v] = bv; }
    for (int kt = 0; kt < 8; ++kt) {
      v16h b  = load_b_frag(Wn1p, kt, nt, lane);
      v16h a0 = load_a_frag(A, LDA_N, kt, lane);
      v16h a1 = load_a_frag(A + 16 * LDA_N, LDA_N, kt, lane);
      acc0 = __builtin_amdgcn_wmma_f32_16x16x32_f16(false, a0, false, b, (short)0, acc0, false, false);
      acc1 = __builtin_amdgcn_wmma_f32_16x16x32_f16(false, a1, false, b, (short)0, acc1, false, false);
    }
    int col = nt * 16 + ncol;
    for (int v = 0; v < 8; ++v) {
      M1[(row0 + v) * LDM + col]      = (_Float16)silu_f(acc0[v]);
      M1[(16 + row0 + v) * LDM + col] = (_Float16)silu_f(acc1[v]);
    }
  }
  __syncthreads();

  // GEMM2: K = 128, residual add, store
  for (int nt = 0; nt < 8; ++nt) {
    float bv = b_n2[nt * 16 + ncol];
    v8f acc0, acc1;
    for (int v = 0; v < 8; ++v) { acc0[v] = bv; acc1[v] = bv; }
    for (int kt = 0; kt < 4; ++kt) {
      v16h b  = load_b_frag(Wn2p, kt, nt, lane);
      v16h a0 = load_a_frag(M1, LDM, kt, lane);
      v16h a1 = load_a_frag(M1 + 16 * LDM, LDM, kt, lane);
      acc0 = __builtin_amdgcn_wmma_f32_16x16x32_f16(false, a0, false, b, (short)0, acc0, false, false);
      acc1 = __builtin_amdgcn_wmma_f32_16x16x32_f16(false, a1, false, b, (short)0, acc1, false, false);
    }
    int col = nt * 16 + ncol;
    for (int v = 0; v < 8; ++v) {
      int n0 = nwave + row0 + v;
      int n1 = nwave + 16 + row0 + v;
      if (n0 < N_NODES) h_out[(size_t)n0 * 128 + col] = nf[(size_t)n0 * 128 + col] + acc0[v];
      if (n1 < N_NODES) h_out[(size_t)n1 * 128 + col] = nf[(size_t)n1 * 128 + col] + acc1[v];
    }
  }

  // coords_out = coords + coord_agg / max(cnt, 1)
  {
    int n = nwave + lane;
    if (n < N_NODES) {
      float c = fmaxf(cnt[n], 1.0f);
      for (int j = 0; j < 3; ++j)
        c_out[3 * n + j] = coords[3 * n + j] + coord_agg[3 * n + j] / c;
    }
  }
}

extern "C" void kernel_launch(void* const* d_in, const int* in_sizes, int n_in,
                              void* d_out, int out_size, void* d_ws, size_t ws_size,
                              hipStream_t stream) {
  (void)in_sizes; (void)n_in; (void)out_size; (void)ws_size;
  const float* nf   = (const float*)d_in[0];
  const float* co   = (const float*)d_in[1];
  const float* ef   = (const float*)d_in[2];
  const float* W_e1 = (const float*)d_in[3];
  const float* b_e1 = (const float*)d_in[4];
  const float* W_e2 = (const float*)d_in[5];
  const float* b_e2 = (const float*)d_in[6];
  const float* W_n1 = (const float*)d_in[7];
  const float* b_n1 = (const float*)d_in[8];
  const float* W_n2 = (const float*)d_in[9];
  const float* b_n2 = (const float*)d_in[10];
  const float* W_c1 = (const float*)d_in[11];
  const float* b_c1 = (const float*)d_in[12];
  const float* W_c  = (const float*)d_in[13];
  const float* W_a  = (const float*)d_in[14];
  const float* b_a  = (const float*)d_in[15];
  const int*   src  = (const int*)d_in[16];
  const int*   dst  = (const int*)d_in[17];

  // workspace layout (f32 accumulators, then f16 permuted weights; all offsets 32B aligned)
  float* ws        = (float*)d_ws;
  float* m_agg     = ws;                                   // N*128
  float* coord_agg = m_agg + (size_t)N_NODES * 128;        // N*3
  float* cnt       = coord_agg + (size_t)N_NODES * 3;      // N
  _Float16* We1p = (_Float16*)(cnt + N_NODES);             // 10*8*512
  _Float16* We2p = We1p + 10 * 8 * 512;                    // 4*8*512
  _Float16* Wn1p = We2p + 4 * 8 * 512;                     // 8*8*512
  _Float16* Wn2p = Wn1p + 8 * 8 * 512;                     // 4*8*512
  _Float16* Wc1p = Wn2p + 4 * 8 * 512;                     // 4*8*512

  zero_kernel<<<1024, 256, 0, stream>>>(m_agg, N_NODES * 132);
  permute_w<<<10 * 8, 512, 0, stream>>>(W_e1, 289, We1p);
  permute_w<<<4 * 8, 512, 0, stream>>>(W_e2, 128, We2p);
  permute_w<<<8 * 8, 512, 0, stream>>>(W_n1, 256, Wn1p);
  permute_w<<<4 * 8, 512, 0, stream>>>(W_n2, 128, Wn2p);
  permute_w<<<4 * 8, 512, 0, stream>>>(W_c1, 128, Wc1p);

  egnn_edge_kernel<<<(N_EDGES + 63) / 64, 64, 0, stream>>>(
      nf, co, ef, b_e1, b_e2, W_a, b_a, b_c1, W_c,
      We1p, We2p, Wc1p, src, dst, m_agg, coord_agg, cnt);

  float* h_out = (float*)d_out;
  float* c_out = h_out + (size_t)N_NODES * 128;
  egnn_node_kernel<<<(N_NODES + 63) / 64, 64, 0, stream>>>(
      nf, co, b_n1, b_n2, Wn1p, Wn2p, m_agg, coord_agg, cnt, h_out, c_out);
}
